// Attention_266287972786
// MI455X (gfx1250) — compile-verified, matrix operand order checked by reading, not run on hardware
//
#include <hip/hip_runtime.h>
#include <hip/hip_bf16.h>
#include <math.h>

typedef __attribute__((ext_vector_type(16))) _Float16 v16h;
typedef __attribute__((ext_vector_type(8)))  float    v8f;

#define B_   64
#define T_   2048
#define RNN_ 1024
#define EMB_ 512
#define ATT_ 128
#define NF_  32
#define KS_  31
#define PAD_ 15
#define TT_  64   // t-tile per block in the WMMA kernel

// Branch-free fast tanh: 1 - 2/(exp(2x)+1).
// exp overflow -> inf -> rcp=0 -> +1 ; exp underflow -> 0 -> rcp(1) -> -1.
__device__ __forceinline__ float tanh_fast(float x)
{
    float e = __expf(2.0f * x);
    return fmaf(-2.0f, __builtin_amdgcn_rcpf(e + 1.0f), 1.0f);
}

// ---------------------------------------------------------------------------
// Kernel 1: processed query  pq[b][a] = dot(h[b,:], Wq[a,:])
// ---------------------------------------------------------------------------
__global__ __launch_bounds__(128) void pq_kernel(
    const float* __restrict__ h, const float* __restrict__ Wq,
    float* __restrict__ pq)
{
    const int b = blockIdx.x;
    const int a = threadIdx.x;            // 128 threads == ATT_
    __shared__ float s_h[RNN_];
    for (int i = threadIdx.x * 4; i < RNN_; i += 128 * 4)
        *(float4*)&s_h[i] = *(const float4*)&h[(size_t)b * RNN_ + i];
    __syncthreads();
    const float* wr = Wq + (size_t)a * RNN_;
    float acc = 0.f;
    #pragma unroll 4
    for (int i = 0; i < RNN_; i += 4) {
        float4 w4 = *(const float4*)&wr[i];
        float4 h4 = *(const float4*)&s_h[i];
        acc = fmaf(w4.x, h4.x, acc);
        acc = fmaf(w4.y, h4.y, acc);
        acc = fmaf(w4.z, h4.z, acc);
        acc = fmaf(w4.w, h4.w, acc);
    }
    pq[b * ATT_ + a] = acc;
}

// ---------------------------------------------------------------------------
// Kernel 2: Wloc f32 -> f16 (128 x 32)
// ---------------------------------------------------------------------------
__global__ void cvt_wloc(const float* __restrict__ Wloc, _Float16* __restrict__ out)
{
    int i = blockIdx.x * 256 + threadIdx.x;
    if (i < ATT_ * NF_) out[i] = (_Float16)Wloc[i];
}

// ---------------------------------------------------------------------------
// Kernel 3: fused conv1d + (loc @ Wloc^T) via WMMA + tanh energy reduction
// grid: (B, T/TT), block: 128 threads (4 waves of 32)
// ---------------------------------------------------------------------------
__global__ __launch_bounds__(128) void conv_energy_kernel(
    const float*    __restrict__ wcat,   // (B, 2, T)
    const float*    __restrict__ pm,     // (B, T, ATT_)
    const float*    __restrict__ pq,     // (B, ATT_)
    const float*    __restrict__ Wconv,  // (NF_, 2, KS_)
    const _Float16* __restrict__ WlocH,  // (ATT_, NF_) f16
    const float*    __restrict__ v,      // (ATT_)
    float*          __restrict__ energies) // (B, T)
{
    const int b   = blockIdx.x;
    const int t0  = blockIdx.y * TT_;
    const int tid = threadIdx.x;

    __shared__ float    s_w[2][TT_ + 2 * PAD_];   // padded conv input window
    __shared__ float    s_conv[NF_ * 2 * KS_];    // conv weights (f, c, k)
    __shared__ _Float16 s_loc[TT_][36];           // loc tile [t][f], pitch 36 halves
    __shared__ _Float16 s_wloc[ATT_ * 36];        // Wloc f16, pitch 36
    __shared__ float    s_pq[ATT_];
    __shared__ float    s_v[ATT_];
    __shared__ float    s_pm[TT_][ATT_];          // staged pm tile (32 KB)

    // ---- stage pm tile with coalesced b128 loads (the 67 MB stream) ----
    {
        const float* pm_blk = pm + ((size_t)b * T_ + t0) * ATT_;
        #pragma unroll
        for (int i = tid * 4; i < TT_ * ATT_; i += 128 * 4)
            *(float4*)(&s_pm[0][0] + i) = *(const float4*)(pm_blk + i);
    }

    // ---- stage remaining inputs into LDS ----
    for (int i = tid; i < 2 * (TT_ + 2 * PAD_); i += 128) {
        int c = i / (TT_ + 2 * PAD_);
        int j = i - c * (TT_ + 2 * PAD_);
        int t = t0 + j - PAD_;
        s_w[c][j] = (t >= 0 && t < T_) ? wcat[((size_t)b * 2 + c) * T_ + t] : 0.f;
    }
    for (int i = tid; i < NF_ * 2 * KS_; i += 128) s_conv[i] = Wconv[i];
    for (int i = tid; i < ATT_ * NF_; i += 128) {
        int r = i >> 5, c = i & 31;
        s_wloc[r * 36 + c] = WlocH[i];
    }
    s_pq[tid] = pq[b * ATT_ + tid];
    s_v[tid]  = v[tid];
    __syncthreads();

    // ---- conv: 32 filters x 64 t = 2048 outputs, 16 per thread ----
    #pragma unroll
    for (int i = 0; i < 16; ++i) {
        int o = tid + 128 * i;
        int f = o >> 6;              // / TT_
        int t = o & (TT_ - 1);
        const float* wf = &s_conv[f * 2 * KS_];
        float acc = 0.f;
        #pragma unroll
        for (int k = 0; k < KS_; ++k) {
            acc = fmaf(wf[k],        s_w[0][t + k], acc);
            acc = fmaf(wf[KS_ + k],  s_w[1][t + k], acc);
        }
        s_loc[t][f] = (_Float16)acc;
    }
    __syncthreads();

    // ---- WMMA: ploc tile = loc(16x32 f16) @ WlocT(32x16 f16), K=32 one shot ----
    const int wave = tid >> 5;        // M-tile index 0..3
    const int lane = tid & 31;
    const int l16  = lane & 15;
    const int hi   = lane >> 4;       // lane group selects K halves / row halves

    // A fragment: row M = wave*16 + l16 for all lanes;
    // lanes 0-15 hold K = 0..7 and 16..23 ; lanes 16-31 hold K = 8..15 and 24..31
    const int arow = wave * 16 + l16;
    v16h afrag;
    {
        const _Float16* ap = &s_loc[arow][hi * 8];
        #pragma unroll
        for (int k = 0; k < 8; ++k) {
            afrag[k]     = ap[k];
            afrag[8 + k] = ap[16 + k];
        }
    }

    float part[8];
    #pragma unroll
    for (int r = 0; r < 8; ++r) part[r] = 0.f;

    const int trow0 = wave * 16 + hi * 8;   // local t row for VGPR r=0

    #pragma unroll
    for (int n = 0; n < ATT_ / 16; ++n) {
        // B fragment: col N = n*16 + l16 ; B[k][N] = Wloc[N][k]
        // lanes 0-15: K = 0..7, 16..23 ; lanes 16-31: K = 8..15, 24..31
        v16h bfrag;
        {
            const _Float16* bp = &s_wloc[(n * 16 + l16) * 36 + hi * 8];
            #pragma unroll
            for (int k = 0; k < 8; ++k) {
                bfrag[k]     = bp[k];
                bfrag[8 + k] = bp[16 + k];
            }
        }
        v8f d = {};
        d = __builtin_amdgcn_wmma_f32_16x16x32_f16(
                /*neg_a=*/false, afrag, /*neg_b=*/false, bfrag,
                /*c_mod=*/(short)0, d, /*reuse_a=*/false, /*reuse_b=*/false);

        // D layout: VGPR r -> row (hi ? r+8 : r), lane l16 -> col
        const int a = n * 16 + l16;
        const float va  = s_v[a];
        const float pqa = s_pq[a];
        #pragma unroll
        for (int r = 0; r < 8; ++r) {
            float x = s_pm[trow0 + r][a] + pqa + d[r];
            part[r] = fmaf(va, tanh_fast(x), part[r]);
        }
    }

    // ---- reduce each row's partial over the 16 lanes of its column group ----
    #pragma unroll
    for (int r = 0; r < 8; ++r) {
        float x = part[r];
        x += __shfl_xor(x, 1, 32);
        x += __shfl_xor(x, 2, 32);
        x += __shfl_xor(x, 4, 32);
        x += __shfl_xor(x, 8, 32);
        if (l16 == 0) {
            int t = t0 + trow0 + r;            // all distinct across lanes/waves
            energies[(size_t)b * T_ + t] = x;
        }
    }
}

// ---------------------------------------------------------------------------
// Kernel 4: masked softmax over T per batch
// ---------------------------------------------------------------------------
__global__ __launch_bounds__(256) void softmax_kernel(
    const float* __restrict__ energies, const unsigned char* __restrict__ mask,
    float* __restrict__ weights)
{
    const int b = blockIdx.x;
    const int tid = threadIdx.x;
    __shared__ float red[256];

    float mx = -INFINITY;
    for (int t = tid; t < T_; t += 256) {
        float e = mask[(size_t)b * T_ + t] ? -INFINITY : energies[(size_t)b * T_ + t];
        mx = fmaxf(mx, e);
    }
    red[tid] = mx; __syncthreads();
    for (int s = 128; s > 0; s >>= 1) {
        if (tid < s) red[tid] = fmaxf(red[tid], red[tid + s]);
        __syncthreads();
    }
    mx = red[0]; __syncthreads();

    float lsum = 0.f;
    for (int t = tid; t < T_; t += 256) {
        float e = mask[(size_t)b * T_ + t] ? -INFINITY : energies[(size_t)b * T_ + t];
        float ex = (e == -INFINITY) ? 0.f : __expf(e - mx);
        weights[(size_t)b * T_ + t] = ex;
        lsum += ex;
    }
    red[tid] = lsum; __syncthreads();
    for (int s = 128; s > 0; s >>= 1) {
        if (tid < s) red[tid] += red[tid + s];
        __syncthreads();
    }
    const float inv = 1.f / red[0];
    __syncthreads();
    for (int t = tid; t < T_; t += 256)
        weights[(size_t)b * T_ + t] *= inv;
}

// ---------------------------------------------------------------------------
// Kernel 5: context[b][e] = sum_t w[b][t] * mem[b][t][e]  (268 MB stream)
// grid: (B), block 128, each lane owns 4 consecutive e (float4)
// ---------------------------------------------------------------------------
__global__ __launch_bounds__(128) void context_kernel(
    const float* __restrict__ mem, const float* __restrict__ weights,
    float* __restrict__ ctx)
{
    const int b = blockIdx.x;
    const int e = threadIdx.x * 4;           // 128 * 4 == EMB_
    __shared__ float s_wt[T_];
    for (int t = threadIdx.x * 4; t < T_; t += 128 * 4)
        *(float4*)&s_wt[t] = *(const float4*)&weights[(size_t)b * T_ + t];
    __syncthreads();

    const float* mp = mem + ((size_t)b * T_) * EMB_ + e;
    float4 acc = {0.f, 0.f, 0.f, 0.f};
    #pragma unroll 4
    for (int t = 0; t < T_; ++t) {
        const float  w = s_wt[t];
        const float4 m = *(const float4*)&mp[(size_t)t * EMB_];
        acc.x = fmaf(w, m.x, acc.x);
        acc.y = fmaf(w, m.y, acc.y);
        acc.z = fmaf(w, m.z, acc.z);
        acc.w = fmaf(w, m.w, acc.w);
    }
    *(float4*)&ctx[(size_t)b * EMB_ + e] = acc;
}

// ---------------------------------------------------------------------------
extern "C" void kernel_launch(void* const* d_in, const int* in_sizes, int n_in,
                              void* d_out, int out_size, void* d_ws, size_t ws_size,
                              hipStream_t stream)
{
    const float* h     = (const float*)d_in[0];   // (B, RNN)
    const float* mem   = (const float*)d_in[1];   // (B, T, EMB)
    const float* pm    = (const float*)d_in[2];   // (B, T, ATT)
    const float* wcat  = (const float*)d_in[3];   // (B, 2, T)
    const unsigned char* mask = (const unsigned char*)d_in[4]; // (B, T) bool
    const float* Wq    = (const float*)d_in[5];   // (ATT, RNN)
    const float* Wconv = (const float*)d_in[6];   // (NF, 2, KS)
    const float* Wloc  = (const float*)d_in[7];   // (ATT, NF)
    const float* v     = (const float*)d_in[8];   // (ATT)

    float* ctx = (float*)d_out;                   // (B, EMB)
    float* wts = ctx + (size_t)B_ * EMB_;         // (B, T)

    char* ws = (char*)d_ws;
    float*    pq_ws  = (float*)ws;                               // B*ATT floats
    float*    en_ws  = (float*)(ws + B_ * ATT_ * sizeof(float)); // B*T floats
    _Float16* wlocH  = (_Float16*)(ws + B_ * ATT_ * sizeof(float)
                                      + (size_t)B_ * T_ * sizeof(float));

    pq_kernel<<<dim3(B_), dim3(128), 0, stream>>>(h, Wq, pq_ws);
    cvt_wloc<<<dim3((ATT_ * NF_ + 255) / 256), dim3(256), 0, stream>>>(Wloc, wlocH);
    conv_energy_kernel<<<dim3(B_, T_ / TT_), dim3(128), 0, stream>>>(
        wcat, pm, pq_ws, Wconv, wlocH, v, en_ws);
    softmax_kernel<<<dim3(B_), dim3(256), 0, stream>>>(en_ws, mask, wts);
    context_kernel<<<dim3(B_), dim3(128), 0, stream>>>(mem, wts, ctx);
}
